// BiLSTMTranslator_75651553952107
// MI455X (gfx1250) — matrix operand order repaired and checked by reading, not compiled, not run on hardware
//
#include <hip/hip_runtime.h>
#include <stddef.h>

// ---------------------------------------------------------------------------
// Types for CDNA5 WMMA (gfx1250, wave32)
// ---------------------------------------------------------------------------
typedef __bf16 bf16;
typedef __attribute__((ext_vector_type(16))) __bf16 bf16x16;
typedef __attribute__((ext_vector_type(8)))  __bf16 bf16x8;
typedef __attribute__((ext_vector_type(8)))  float  floatx8;
typedef __attribute__((ext_vector_type(4)))  unsigned int uintx4;
typedef __attribute__((ext_vector_type(8)))  unsigned int uintx8;

union ABfrag { bf16x16 v; bf16x8 h[2]; };

__device__ __forceinline__ floatx8 zero8() {
  floatx8 z = {0.f,0.f,0.f,0.f,0.f,0.f,0.f,0.f};
  return z;
}

__device__ __forceinline__ floatx8 wmma_bf16(bf16x16 a, bf16x16 b, floatx8 c) {
  // v_wmma_f32_16x16x32_bf16  (neg_a, A, neg_b, B, c_mod, C, reuse_a, reuse_b)
  return __builtin_amdgcn_wmma_f32_16x16x32_bf16(false, a, false, b, (short)0, c, false, false);
}

__device__ __forceinline__ float sigmoidf_(float x) { return 1.f / (1.f + __expf(-x)); }
__device__ __forceinline__ float tanhf_(float x)    { return 1.f - 2.f / (__expf(2.f * x) + 1.f); }

// Fragment layouts used by this code (A/B k order is contiguous, not the ISA table order):
//  A 16x32 bf16: lane L holds row M=L&15; lanes 0-15: K 0..15; lanes 16-31: K 16..31
//  B 32x16 bf16: lane L holds col N=L&15; lanes 0-15: K 0..15; lanes 16-31: K 16..31
//  C 16x16 f32 : vgpr r, lane L -> (M = r + (L>>4)*8, N = L&15)
__device__ __forceinline__ bf16x16 load_a_frag(const bf16* row_base, int hi8) {
  ABfrag au;
  au.h[0] = *(const bf16x8*)(row_base + 2 * hi8);
  au.h[1] = *(const bf16x8*)(row_base + 2 * hi8 + 8);
  return au.v;
}

// Problem constants
#define BB   16
#define SS   128
#define TT   64
#define EE   512
#define HH   512
#define GG   2048     // 4*H
#define VZH  32000

// ---------------------------------------------------------------------------
// Small prep kernels
// ---------------------------------------------------------------------------
__global__ void k_f32_to_bf16(const float* __restrict__ src, bf16* __restrict__ dst, long n) {
  long i = (long)blockIdx.x * blockDim.x + threadIdx.x;
  long stride = (long)gridDim.x * blockDim.x;
  for (; i < n; i += stride) dst[i] = (bf16)src[i];
}

__global__ void k_split_wattn(const float* __restrict__ W, bf16* __restrict__ wa_h,
                              bf16* __restrict__ wa_e) {
  int i = blockIdx.x * blockDim.x + threadIdx.x;  // over 512*512
  if (i >= HH * HH) return;
  int n = i >> 9, k = i & (HH - 1);
  wa_h[i] = (bf16)W[n * (2 * HH) + k];
  wa_e[i] = (bf16)W[n * (2 * HH) + HH + k];
}

// x_bf16[(s*16+b)*512 + e] = bf16(en_emb[src[b,s], e])
__global__ void k_embed_src(const int* __restrict__ src, const float* __restrict__ emb,
                            bf16* __restrict__ x) {
  long i = (long)blockIdx.x * blockDim.x + threadIdx.x;  // S*B*E = 1M
  if (i >= (long)SS * BB * EE) return;
  int e = (int)(i & (EE - 1));
  int b = (int)((i >> 9) & (BB - 1));
  int s = (int)(i >> 13);
  x[i] = (bf16)emb[(long)src[b * SS + s] * EE + e];
}

// ---------------------------------------------------------------------------
// Generic bf16 WMMA GEMM: C[M,N] = A[M,K] * W[N,K]^T (+ b0 + b1)
// Each wave computes 4 adjacent 16x16 N-tiles (A-fragment amortized 4x).
// Output f32 (Cf) or bf16 (Cb). Ntiles must be a multiple of 4.
// ---------------------------------------------------------------------------
__global__ void k_gemm_bf16(const bf16* __restrict__ A, const bf16* __restrict__ W,
                            const float* __restrict__ b0, const float* __restrict__ b1,
                            float* __restrict__ Cf, bf16* __restrict__ Cb,
                            int Mtiles, int Ntiles, int K) {
  int wave = (blockIdx.x * blockDim.x + threadIdx.x) >> 5;
  int lane = threadIdx.x & 31;
  int Ngrp = Ntiles >> 2;
  int tm = wave / Ngrp, tg = wave - tm * Ngrp;
  if (tm >= Mtiles) return;
  int tn0 = tg << 2;
  int hi = lane >> 4, lo = lane & 15, hi8 = hi << 3;
  int N = Ntiles << 4;

  const bf16* arow  = A + (size_t)(tm * 16 + lo) * K;
  const bf16* wrow0 = W + (size_t)(tn0 * 16 + lo) * K + hi * 16;
  const size_t wtile = (size_t)16 * K;

  floatx8 acc[4];
  for (int j = 0; j < 4; ++j) acc[j] = zero8();
  for (int k0 = 0; k0 < K; k0 += 32) {
    bf16x16 af = load_a_frag(arow + k0, hi8);
    for (int j = 0; j < 4; ++j) {
      bf16x16 bf = *(const bf16x16*)(wrow0 + (size_t)j * wtile + k0);
      acc[j] = wmma_bf16(af, bf, acc[j]);
    }
  }
  for (int j = 0; j < 4; ++j) {
    int n = (tn0 + j) * 16 + lo;
    float bias = (b0 ? b0[n] : 0.f) + (b1 ? b1[n] : 0.f);
    for (int r = 0; r < 8; ++r) {
      int m = tm * 16 + r + hi8;
      float val = acc[j][r] + bias;
      if (Cf) Cf[(size_t)m * N + n] = val;
      else    Cb[(size_t)m * N + n] = (bf16)val;
    }
  }
}

// ---------------------------------------------------------------------------
// Encoder LSTM scan (one direction). One 1024-thread block (32 waves).
// Wave w owns hidden cols [16w,16w+16) across all 4 gates -> gates & c stay
// in WMMA accumulator registers; h round-trips through LDS as bf16.
// ---------------------------------------------------------------------------
__global__ void __launch_bounds__(1024)
k_lstm_dir(const float* __restrict__ xg, const bf16* __restrict__ Whh,
           float* __restrict__ enc_out, bf16* __restrict__ hT, float* __restrict__ cT,
           int reverse, int accumulate) {
  __shared__ bf16 hbuf[BB * HH];  // 16 KB
  int tid = threadIdx.x, lane = tid & 31, w = tid >> 5;
  int hi = lane >> 4, lo = lane & 15, hi8 = hi << 3;

  for (int i = tid; i < BB * HH; i += 1024) hbuf[i] = (bf16)0.f;
  floatx8 c = zero8();
  __syncthreads();

  for (int step = 0; step < SS; ++step) {
    int s = reverse ? (SS - 1 - step) : step;

    floatx8 g4[4];
    for (int g = 0; g < 4; ++g) g4[g] = zero8();
    for (int k0 = 0; k0 < HH; k0 += 32) {
      bf16x16 af = load_a_frag(&hbuf[lo * HH + k0], hi8);
      for (int g = 0; g < 4; ++g) {
        const bf16* bp = Whh + (size_t)(g * HH + w * 16 + lo) * HH + k0 + hi * 16;
        g4[g] = wmma_bf16(af, *(const bf16x16*)bp, g4[g]);
      }
    }
    __syncthreads();  // all hbuf reads done before rewrite

    for (int r = 0; r < 8; ++r) {
      int m = r + hi8;             // batch row
      int col = w * 16 + lo;       // hidden column
      const float* xr = xg + (size_t)(s * BB + m) * GG;
      float iv = sigmoidf_(g4[0][r] + xr[0 * HH + col]);
      float fv = sigmoidf_(g4[1][r] + xr[1 * HH + col]);
      float gv = tanhf_   (g4[2][r] + xr[2 * HH + col]);
      float ov = sigmoidf_(g4[3][r] + xr[3 * HH + col]);
      float cv = fv * c[r] + iv * gv;
      c[r] = cv;
      float hv = ov * tanhf_(cv);
      hbuf[m * HH + col] = (bf16)hv;
      float* eo = enc_out + (size_t)(m * SS + s) * HH + col;
      if (accumulate) *eo = *eo + hv; else *eo = hv;
    }
    __syncthreads();  // hbuf fully updated before next step's reads
  }
  for (int r = 0; r < 8; ++r) {
    int m = r + hi8, col = w * 16 + lo;
    cT[m * HH + col] = c[r];
    hT[m * HH + col] = hbuf[m * HH + col];
  }
}

// ---------------------------------------------------------------------------
// Decoder: attention + LSTM cell, 64 steps in one 1024-thread block.
// Target-embedding rows are fetched with the Tensor Data Mover in gather mode
// (16 token-indexed rows of zh_emb -> LDS), overlapped with the attention
// math and fenced with s_wait_tensorcnt.
// ---------------------------------------------------------------------------
__global__ void __launch_bounds__(1024)
k_decoder(const int* __restrict__ tgt, const float* __restrict__ zh_emb,
          const bf16* __restrict__ encp, const float* __restrict__ enc_out,
          const bf16* __restrict__ WaH, const float* __restrict__ battn,
          const float* __restrict__ vv,
          const bf16* __restrict__ Wihd, const bf16* __restrict__ Whhd,
          const float* __restrict__ bihd, const float* __restrict__ bhhd,
          const bf16* __restrict__ hT, const float* __restrict__ cT,
          bf16* __restrict__ dec_hc) {
  __shared__ bf16  hbuf[BB * HH];       // 16 KB  current h (bf16)
  __shared__ bf16  xin[BB * 1024];      // 32 KB  [emb | context]; aliased as hwa early in step
  __shared__ float sc[BB * SS];         // 8 KB   scores / attention
  __shared__ float embstage[BB * EE];   // 32 KB  TDM gather landing zone (f32 rows)
  bf16* hwabuf = xin;                   // hwa (16x512) lives here before xin is built

  int tid = threadIdx.x, lane = tid & 31, w = tid >> 5;
  int hi = lane >> 4, lo = lane & 15, hi8 = hi << 3;

  // init h, c from backward-encoder final state
  for (int i = tid; i < BB * HH; i += 1024) hbuf[i] = hT[i];
  floatx8 c = zero8();
  for (int r = 0; r < 8; ++r) c[r] = cT[(r + hi8) * HH + w * 16 + lo];
  __syncthreads();

  // Constant parts of the TDM descriptor (D#), per CDNA5 ISA
  unsigned long ga = (unsigned long)(const void*)zh_emb;
  unsigned lds_off = (unsigned)(size_t)(void*)embstage;  // flat addr low 32 = LDS offset
  uintx4 g0;
  g0[0] = (1u << 31) | 1u;                    // gather_mode=1, 16-bit indices, count=1
  g0[1] = lds_off;                            // lds_addr
  g0[2] = (unsigned)ga;                       // global_addr[31:0]
  g0[3] = (unsigned)((ga >> 32) & 0x1FFFFFFu) | (2u << 30);  // addr[56:32] | type=2
  uintx8 g1;
  g1[0] = 2u << 16;                           // data_size=2 (4 bytes), wg_mask=0
  g1[1] = (unsigned)(EE & 0xFFFF) << 16;      // tensor_dim0 = 512 (bits 79:48, low half)
  g1[2] = (unsigned)(VZH & 0xFFFF) << 16;     // dim0 hi=0 | tensor_dim1 = 32000 (low half)
  g1[3] = (unsigned)(EE & 0xFFFF) << 16;      // dim1 hi=0 | tile_dim0 = 512
  g1[4] = (unsigned)BB;                       // tile_dim1 = 16 valid indices
  g1[5] = (unsigned)EE;                       // tensor_dim0_stride = 512
  g1[6] = 0u;
  g1[7] = 0u;

  int tok = tgt[(lane & 15) * TT];            // lane b holds token for batch b (step 0)

  for (int t = 0; t < TT; ++t) {
    // (a0) wave 0: pack the 16 token row-indices into D# groups 2/3, issue TDM gather
    if (w == 0) {
      uintx4 g2, g3;
      for (int j = 0; j < 4; ++j) {
        g2[j] = ((unsigned)__builtin_amdgcn_readlane(tok, 2 * j) & 0xFFFFu) |
                ((unsigned)__builtin_amdgcn_readlane(tok, 2 * j + 1) << 16);
        g3[j] = ((unsigned)__builtin_amdgcn_readlane(tok, 2 * j + 8) & 0xFFFFu) |
                ((unsigned)__builtin_amdgcn_readlane(tok, 2 * j + 9) << 16);
      }
      asm volatile("tensor_load_to_lds %0, %1, %2, %3"
                   :: "s"(g0), "s"(g1), "s"(g2), "s"(g3)
                   : "memory");
    }
    // prefetch next step's tokens (uniform per-lane load, hides latency)
    if (t + 1 < TT) tok = tgt[(lane & 15) * TT + (t + 1)];

    // (a) hwa = h @ Wa_h^T + battn   (32 N-tiles -> one per wave) — overlaps the TDM
    floatx8 ha = zero8();
    for (int k0 = 0; k0 < HH; k0 += 32) {
      bf16x16 af = load_a_frag(&hbuf[lo * HH + k0], hi8);
      const bf16* bp = WaH + (size_t)(w * 16 + lo) * HH + k0 + hi * 16;
      ha = wmma_bf16(af, *(const bf16x16*)bp, ha);
    }
    for (int r = 0; r < 8; ++r) {
      int m = r + hi8, col = w * 16 + lo;
      hwabuf[m * HH + col] = (bf16)(ha[r] + battn[col]);
    }
    __syncthreads();

    // (b) score[b,s] = v . tanh(enc_proj[b,s,:] + hwa[b,:])  (wave per (b,s) pair)
    for (int i = 0; i < 64; ++i) {
      int p = w + 32 * i;                     // 0..2047
      int b = p >> 7, s = p & (SS - 1);
      const bf16* ep = encp + (size_t)(b * SS + s) * HH;
      const bf16* hp = hwabuf + b * HH;
      float part = 0.f;
      for (int j = 0; j < 16; ++j) {
        int h = j * 32 + lane;
        part += vv[h] * tanhf_((float)ep[h] + (float)hp[h]);
      }
      for (int off = 16; off; off >>= 1) part += __shfl_xor(part, off, 32);
      if (lane == 0) sc[b * SS + s] = part;
    }
    __syncthreads();

    // (c) softmax over s (wave per batch row)
    if (w < BB) {
      int b = w;
      float vals[4], vmax = -1e30f;
      for (int i = 0; i < 4; ++i) { vals[i] = sc[b * SS + lane + i * 32]; vmax = fmaxf(vmax, vals[i]); }
      for (int off = 16; off; off >>= 1) vmax = fmaxf(vmax, __shfl_xor(vmax, off, 32));
      float sum = 0.f;
      for (int i = 0; i < 4; ++i) { vals[i] = __expf(vals[i] - vmax); sum += vals[i]; }
      for (int off = 16; off; off >>= 1) sum += __shfl_xor(sum, off, 32);
      float inv = 1.f / sum;
      for (int i = 0; i < 4; ++i) sc[b * SS + lane + i * 32] = vals[i] * inv;
    }
    // fence the TDM gather before anyone reads embstage
    if (w == 0) __builtin_amdgcn_s_wait_tensorcnt(0);
    __syncthreads();

    // (d) context + build xin = [emb | context] (overwrites hwa alias)
    for (int p = tid; p < BB * HH; p += 1024) {
      int b = p >> 9, h = p & (HH - 1);
      const float* eo = enc_out + (size_t)(b * SS) * HH + h;
      const float* at = sc + b * SS;
      float ctx = 0.f;
      for (int s = 0; s < SS; ++s) ctx += at[s] * eo[(size_t)s * HH];
      xin[b * 1024 + HH + h] = (bf16)ctx;
      dec_hc[((size_t)t * BB + b) * 1024 + HH + h] = (bf16)ctx;
      xin[b * 1024 + h] = (bf16)embstage[b * EE + h];   // TDM-gathered embedding row
    }
    __syncthreads();

    // (e) gates = xin @ Wih_d^T (K=1024) + h @ Whh_d^T (K=512)
    floatx8 g4[4];
    for (int g = 0; g < 4; ++g) g4[g] = zero8();
    for (int k0 = 0; k0 < 1024; k0 += 32) {
      bf16x16 af = load_a_frag(&xin[lo * 1024 + k0], hi8);
      for (int g = 0; g < 4; ++g) {
        const bf16* bp = Wihd + (size_t)(g * HH + w * 16 + lo) * 1024 + k0 + hi * 16;
        g4[g] = wmma_bf16(af, *(const bf16x16*)bp, g4[g]);
      }
    }
    for (int k0 = 0; k0 < HH; k0 += 32) {
      bf16x16 af = load_a_frag(&hbuf[lo * HH + k0], hi8);
      for (int g = 0; g < 4; ++g) {
        const bf16* bp = Whhd + (size_t)(g * HH + w * 16 + lo) * HH + k0 + hi * 16;
        g4[g] = wmma_bf16(af, *(const bf16x16*)bp, g4[g]);
      }
    }
    __syncthreads();  // xin/hbuf reads complete before rewrites

    // (f) LSTM cell update; store h into hbuf + dec_hc
    for (int r = 0; r < 8; ++r) {
      int m = r + hi8, col = w * 16 + lo;
      float iv = sigmoidf_(g4[0][r] + bihd[0 * HH + col] + bhhd[0 * HH + col]);
      float fv = sigmoidf_(g4[1][r] + bihd[1 * HH + col] + bhhd[1 * HH + col]);
      float gv = tanhf_   (g4[2][r] + bihd[2 * HH + col] + bhhd[2 * HH + col]);
      float ov = sigmoidf_(g4[3][r] + bihd[3 * HH + col] + bhhd[3 * HH + col]);
      float cv = fv * c[r] + iv * gv;
      c[r] = cv;
      float hv = ov * tanhf_(cv);
      hbuf[m * HH + col] = (bf16)hv;
      dec_hc[((size_t)t * BB + m) * 1024 + col] = (bf16)hv;
    }
    __syncthreads();
  }
}

// ---------------------------------------------------------------------------
// Logits: out[b,t,v] = dec_hc[t,b,:] . Wout[v,:] + bout[v]
// M = T*B = 1024 rows (m = t*16+b), N = 32000, K = 1024.
// 4 N-tiles per wave; Wout (65.5 MB bf16) is L2-resident across the kernel.
// ---------------------------------------------------------------------------
__global__ void k_logits(const bf16* __restrict__ A, const bf16* __restrict__ Wb,
                         const float* __restrict__ bout, float* __restrict__ out) {
  const int Mtiles = (TT * BB) / 16;   // 64
  const int Ngrp   = (VZH / 16) / 4;   // 500 groups of 4 tiles
  int wave = (blockIdx.x * blockDim.x + threadIdx.x) >> 5;
  int lane = threadIdx.x & 31;
  int tm = wave / Ngrp, tg = wave - tm * Ngrp;
  if (tm >= Mtiles) return;
  int tn0 = tg << 2;
  int hi = lane >> 4, lo = lane & 15, hi8 = hi << 3;

  const bf16* arow  = A + (size_t)(tm * 16 + lo) * 1024;
  const bf16* wrow0 = Wb + (size_t)(tn0 * 16 + lo) * 1024 + hi * 16;
  const size_t wtile = (size_t)16 * 1024;

  floatx8 acc[4];
  for (int j = 0; j < 4; ++j) acc[j] = zero8();
  for (int k0 = 0; k0 < 1024; k0 += 32) {
    __builtin_prefetch(wrow0 + k0 + 128, 0, 1);   // global_prefetch_b8 on weight stream
    bf16x16 af = load_a_frag(arow + k0, hi8);
    for (int j = 0; j < 4; ++j) {
      bf16x16 bf = *(const bf16x16*)(wrow0 + (size_t)j * wtile + k0);
      acc[j] = wmma_bf16(af, bf, acc[j]);
    }
  }
  for (int j = 0; j < 4; ++j) {
    int n = (tn0 + j) * 16 + lo;
    float bias = bout[n];
    for (int r = 0; r < 8; ++r) {
      int m = tm * 16 + r + hi8;
      int t = m >> 4, b = m & 15;
      out[((size_t)b * TT + t) * VZH + n] = acc[j][r] + bias;
    }
  }
}

// ---------------------------------------------------------------------------
// Host launcher
// ---------------------------------------------------------------------------
extern "C" void kernel_launch(void* const* d_in, const int* in_sizes, int n_in,
                              void* d_out, int out_size, void* d_ws, size_t ws_size,
                              hipStream_t stream) {
  (void)in_sizes; (void)n_in; (void)out_size; (void)ws_size;
  const int*   src    = (const int*)  d_in[0];
  const int*   tgt    = (const int*)  d_in[1];
  const float* en_emb = (const float*)d_in[2];
  const float* zh_emb = (const float*)d_in[3];
  const float* Wih_f  = (const float*)d_in[4];
  const float* Whh_f  = (const float*)d_in[5];
  const float* bih_f  = (const float*)d_in[6];
  const float* bhh_f  = (const float*)d_in[7];
  const float* Wih_b  = (const float*)d_in[8];
  const float* Whh_b  = (const float*)d_in[9];
  const float* bih_b  = (const float*)d_in[10];
  const float* bhh_b  = (const float*)d_in[11];
  const float* Wih_d  = (const float*)d_in[12];
  const float* Whh_d  = (const float*)d_in[13];
  const float* bih_d  = (const float*)d_in[14];
  const float* bhh_d  = (const float*)d_in[15];
  const float* Wattn  = (const float*)d_in[16];
  const float* battn  = (const float*)d_in[17];
  const float* vvec   = (const float*)d_in[18];
  const float* Wout   = (const float*)d_in[19];
  const float* bout   = (const float*)d_in[20];
  float* out = (float*)d_out;

  // Workspace carve-up (256B aligned)
  char* base = (char*)d_ws;
  size_t off = 0;
  auto carve = [&](size_t bytes) -> char* {
    char* p = base + off;
    off += (bytes + 255) & ~(size_t)255;
    return p;
  };
  bf16*  x_bf    = (bf16*) carve((size_t)SS * BB * EE * 2);      // 2 MB
  bf16*  wihf_b  = (bf16*) carve((size_t)GG * EE * 2);           // 2 MB
  bf16*  whhf_b  = (bf16*) carve((size_t)GG * HH * 2);
  bf16*  wihb_b  = (bf16*) carve((size_t)GG * EE * 2);
  bf16*  whhb_b  = (bf16*) carve((size_t)GG * HH * 2);
  bf16*  wihd_b  = (bf16*) carve((size_t)GG * 1024 * 2);         // 4 MB
  bf16*  whhd_b  = (bf16*) carve((size_t)GG * HH * 2);
  bf16*  wah_b   = (bf16*) carve((size_t)HH * HH * 2);
  bf16*  wae_b   = (bf16*) carve((size_t)HH * HH * 2);
  bf16*  wout_b  = (bf16*) carve((size_t)VZH * 1024 * 2);        // 65.5 MB
  float* xg_f    = (float*)carve((size_t)SS * BB * GG * 4);      // 16 MB
  float* xg_b    = (float*)carve((size_t)SS * BB * GG * 4);      // 16 MB
  float* enc_out = (float*)carve((size_t)BB * SS * HH * 4);      // 4 MB
  bf16*  enc_bf  = (bf16*) carve((size_t)BB * SS * HH * 2);      // 2 MB
  bf16*  encp_bf = (bf16*) carve((size_t)BB * SS * HH * 2);      // 2 MB
  bf16*  hT      = (bf16*) carve((size_t)BB * HH * 2);
  float* cT      = (float*)carve((size_t)BB * HH * 4);
  bf16*  dec_hc  = (bf16*) carve((size_t)TT * BB * 1024 * 2);    // 2 MB

  // --- weight / activation conversion to bf16 ---
  hipLaunchKernelGGL(k_f32_to_bf16, dim3(2048), dim3(256), 0, stream, Wih_f, wihf_b, (long)GG * EE);
  hipLaunchKernelGGL(k_f32_to_bf16, dim3(2048), dim3(256), 0, stream, Whh_f, whhf_b, (long)GG * HH);
  hipLaunchKernelGGL(k_f32_to_bf16, dim3(2048), dim3(256), 0, stream, Wih_b, wihb_b, (long)GG * EE);
  hipLaunchKernelGGL(k_f32_to_bf16, dim3(2048), dim3(256), 0, stream, Whh_b, whhb_b, (long)GG * HH);
  hipLaunchKernelGGL(k_f32_to_bf16, dim3(4096), dim3(256), 0, stream, Wih_d, wihd_b, (long)GG * 1024);
  hipLaunchKernelGGL(k_f32_to_bf16, dim3(2048), dim3(256), 0, stream, Whh_d, whhd_b, (long)GG * HH);
  hipLaunchKernelGGL(k_f32_to_bf16, dim3(8192), dim3(256), 0, stream, Wout,  wout_b, (long)VZH * 1024);
  hipLaunchKernelGGL(k_split_wattn, dim3((HH * HH + 255) / 256), dim3(256), 0, stream, Wattn, wah_b, wae_b);
  hipLaunchKernelGGL(k_embed_src,   dim3(((long)SS * BB * EE + 255) / 256), dim3(256), 0, stream,
                     src, en_emb, x_bf);

  // --- encoder input projections: xg = x @ Wih^T + bih + bhh  (M=2048,N=2048,K=512) ---
  {
    int Mt = (SS * BB) / 16, Nt = GG / 16;
    int blocks = (Mt * (Nt / 4) + 7) / 8;
    hipLaunchKernelGGL(k_gemm_bf16, dim3(blocks), dim3(256), 0, stream,
                       x_bf, wihf_b, bih_f, bhh_f, xg_f, (bf16*)nullptr, Mt, Nt, EE);
    hipLaunchKernelGGL(k_gemm_bf16, dim3(blocks), dim3(256), 0, stream,
                       x_bf, wihb_b, bih_b, bhh_b, xg_b, (bf16*)nullptr, Mt, Nt, EE);
  }

  // --- recurrent scans (forward writes enc_out, backward accumulates; bwd final state kept) ---
  hipLaunchKernelGGL(k_lstm_dir, dim3(1), dim3(1024), 0, stream,
                     xg_f, whhf_b, enc_out, hT, cT, 0 /*fwd*/, 0 /*write*/);
  hipLaunchKernelGGL(k_lstm_dir, dim3(1), dim3(1024), 0, stream,
                     xg_b, whhb_b, enc_out, hT, cT, 1 /*rev*/, 1 /*accum*/);

  // --- enc_out -> bf16, then enc_proj = enc_out @ Wa_e^T (bf16 out) ---
  hipLaunchKernelGGL(k_f32_to_bf16, dim3(2048), dim3(256), 0, stream,
                     enc_out, enc_bf, (long)BB * SS * HH);
  {
    int Mt = (BB * SS) / 16, Nt = HH / 16;
    int blocks = (Mt * (Nt / 4) + 7) / 8;
    hipLaunchKernelGGL(k_gemm_bf16, dim3(blocks), dim3(256), 0, stream,
                       enc_bf, wae_b, (const float*)nullptr, (const float*)nullptr,
                       (float*)nullptr, encp_bf, Mt, Nt, HH);
  }

  // --- decoder scan: attention + LSTM (with TDM embedding gather) ---
  hipLaunchKernelGGL(k_decoder, dim3(1), dim3(1024), 0, stream,
                     tgt, zh_emb, encp_bf, enc_out, wah_b, battn, vvec,
                     wihd_b, whhd_b, bih_d, bhh_d, hT, cT, dec_hc);

  // --- logits GEMM: (T*B=1024) x 32000, K=1024, 4 N-tiles per wave ---
  {
    int waves = ((TT * BB) / 16) * ((VZH / 16) / 4);  // 64 * 500
    int blocks = (waves + 7) / 8;
    hipLaunchKernelGGL(k_logits, dim3(blocks), dim3(256), 0, stream,
                       dec_hc, wout_b, bout, out);
  }
}
